// MultiHeadAttentionWithThreshold_46566035423440
// MI455X (gfx1250) — matrix operand order repaired and checked by reading, not compiled, not run on hardware
//
#include <hip/hip_runtime.h>
#include <hip/hip_bf16.h>
#include <math.h>

typedef _Float16 half_t;
typedef __attribute__((ext_vector_type(16))) _Float16 v16h;
typedef __attribute__((ext_vector_type(8)))  _Float16 v8h;
typedef __attribute__((ext_vector_type(8)))  float    v8f;

#define D_MODEL 1024
#define NHEAD   16
#define DK      64
#define SEQ     2048
#define BATCH   2
#define MROWS   (BATCH * SEQ)   // 4096
#define NEGINF  (-1e9f)

#define GN_WAVE  64             // N columns per wave (4 x 16 tiles)
#define GN_BLOCK 512            // 8 waves * 64
#define KCHUNK   64             // K step staged through LDS (2 wmma k-steps)

static __device__ inline v16h cat8(v8h a, v8h b) {
    return __builtin_shufflevector(a, b, 0,1,2,3,4,5,6,7,8,9,10,11,12,13,14,15);
}

// ---------------------------------------------------------------------------
// f32 -> f16 conversion
// ---------------------------------------------------------------------------
__global__ void cvt_f32_f16(const float* __restrict__ in, half_t* __restrict__ out, int n) {
    int i = blockIdx.x * blockDim.x + threadIdx.x;
    if (i < n) out[i] = (half_t)in[i];
}

// ---------------------------------------------------------------------------
// Y[M,N] = A[M,K] @ W[N,K]^T + bias[N]     (A, W are f16; accum f32)
// mode 0: f16 out, head-split   [B,H,S,DK]
// mode 1: f16 out, transposed   [B,H,DK,S]   (for V)
// mode 2: f32 out, row-major    [M,N]        (final output)
//
// Block = 256 threads (8 waves); block tile = 16(M) x 512(N); each wave owns
// a 16x64 strip (one A fragment reused across 4 WMMAs). The 16 x KCHUNK A
// tile is async-copied into double-buffered LDS with
// GLOBAL_LOAD_ASYNC_TO_LDS_B128 (ASYNCcnt) and shared by all 8 waves.
// Per chunk: bulk-load all fragments, then 8 back-to-back WMMAs.
// ---------------------------------------------------------------------------
__global__ __launch_bounds__(256) void gemm_xwT_kernel(
    const half_t* __restrict__ A, const half_t* __restrict__ W,
    const float* __restrict__ bias, half_t* __restrict__ out_h,
    float* __restrict__ out_f, int K, int N, int mode)
{
    __shared__ half_t sA[2 * 16 * KCHUNK];   // 4 KB double buffer

    const int tid   = threadIdx.x;
    const int lane  = tid & 31;
    const int wave  = tid >> 5;
    const int l15   = lane & 15;
    const int hi    = lane >> 4;
    const int tileM = blockIdx.x * 16;
    const int tileN = blockIdx.y * GN_BLOCK + wave * GN_WAVE;

    // threads 0..127: one 16B async copy each per K-chunk (16 rows x 8 segs)
    const int crow = tid >> 3;          // row 0..15 (tid<128)
    const int cseg = tid & 7;           // 16B segment 0..7
    const half_t* agsrc = A + (size_t)(tileM + crow) * K + cseg * 8;

    auto issue = [&](int kk, int buf) {
        if (tid < 128) {
            unsigned int ldsa = (unsigned int)(reinterpret_cast<uintptr_t>(
                sA + buf * (16 * KCHUNK) + crow * KCHUNK + cseg * 8));
            unsigned long long ga =
                (unsigned long long)(reinterpret_cast<uintptr_t>(agsrc + kk));
            asm volatile("global_load_async_to_lds_b128 %0, %1, off"
                         :: "v"(ldsa), "v"(ga) : "memory");
        }
    };

    v8f c[4];
    #pragma unroll
    for (int u = 0; u < 4; ++u)
        #pragma unroll
        for (int r = 0; r < 8; ++r) c[u][r] = 0.f;

    auto compute = [&](int ch, int buf) {
        const half_t* abase = sA + buf * (16 * KCHUNK) + l15 * KCHUNK;
        v16h af[2], bf[2][4];
        // A fragments from LDS: per-lane row, K split (0-15)/(16-31)
        #pragma unroll
        for (int t = 0; t < 2; ++t) {
            const int ka = t * 32 + (hi ? 16 : 0);
            af[t] = cat8(*(const v8h*)(abase + ka),
                         *(const v8h*)(abase + ka + 8));
        }
        // B fragments from global: per-lane column, 16 contiguous K values
        #pragma unroll
        for (int t = 0; t < 2; ++t) {
            const int kb = ch * KCHUNK + t * 32 + (hi ? 16 : 0);
            #pragma unroll
            for (int u = 0; u < 4; ++u)
                bf[t][u] = *(const v16h*)(W + (size_t)(tileN + u * 16 + l15) * K + kb);
        }
        // 8 back-to-back WMMAs
        #pragma unroll
        for (int t = 0; t < 2; ++t)
            #pragma unroll
            for (int u = 0; u < 4; ++u)
                c[u] = __builtin_amdgcn_wmma_f32_16x16x32_f16(false, af[t], false,
                                                              bf[t][u], (short)0,
                                                              c[u], false, false);
    };

    const int nc = K / KCHUNK;
    issue(0, 0);
    for (int ch = 0; ch < nc - 1; ++ch) {
        issue((ch + 1) * KCHUNK, (ch + 1) & 1);
        // async completes in order: <=1 outstanding => chunk ch has landed
        asm volatile("s_wait_asynccnt 0x1" ::: "memory");
        __syncthreads();                 // chunk ch visible to all waves
        compute(ch, ch & 1);
        __syncthreads();                 // reads done before chunk ch+2 overwrite
    }
    asm volatile("s_wait_asynccnt 0x0" ::: "memory");
    __syncthreads();
    compute(nc - 1, (nc - 1) & 1);

    #pragma unroll
    for (int u = 0; u < 4; ++u) {
        const int colg = tileN + u * 16 + l15;
        const float bv = bias[colg];
        #pragma unroll
        for (int r = 0; r < 8; ++r) {
            const int rowg = tileM + r + (hi ? 8 : 0);
            const float val = c[u][r] + bv;
            if (mode == 2) {
                out_f[(size_t)rowg * N + colg] = val;
            } else {
                const int b = rowg >> 11, s = rowg & (SEQ - 1);
                const int h = colg >> 6,  d = colg & (DK - 1);
                size_t idx;
                if (mode == 0) idx = (((size_t)(b * NHEAD + h) * SEQ) + s) * DK + d;
                else           idx = (((size_t)(b * NHEAD + h) * DK)  + d) * SEQ + s;
                out_h[idx] = (half_t)val;
            }
        }
    }
}

// ---------------------------------------------------------------------------
// Fused thresholded attention, flash-style streaming softmax.
// One wave handles a 16-query tile of one (b,h); streams keys 32 at a time.
//   S = (Q K^T)/8 ; S = (S<=thr ? -1e9 : S) ; online softmax ; O += P V
// P is produced in C-layout and transposed to A-layout through LDS.
// Output written f16 in merged-head layout [B,S,H*DK] for the final GEMM.
// ---------------------------------------------------------------------------
__global__ __launch_bounds__(128) void attn_kernel(
    const half_t* __restrict__ Qh, const half_t* __restrict__ Kh,
    const half_t* __restrict__ Vt, const float* __restrict__ raw_thr,
    half_t* __restrict__ O)
{
    __shared__ half_t lds[4 * 16 * 32];   // per-wave 16x32 P staging (f16)

    const int lane = threadIdx.x & 31;
    const int wave = threadIdx.x >> 5;
    const int l15  = lane & 15;
    const int hi   = lane >> 4;
    const int h    = blockIdx.y;
    const int b    = blockIdx.z;
    const int q0   = (blockIdx.x * 4 + wave) * 16;

    const float raw = raw_thr[h];
    const float thr = (raw > 20.f) ? raw : log1pf(__expf(raw));

    const size_t bh = (size_t)(b * NHEAD + h);

    // Q A-fragments for d-chunks [0,32) and [32,64)
    const half_t* qrow = Qh + (bh * SEQ + q0 + l15) * DK;
    v16h qa[2];
    #pragma unroll
    for (int f = 0; f < 2; ++f) {
        const int ka = f * 32 + (hi ? 16 : 0);
        qa[f] = cat8(*(const v8h*)(qrow + ka), *(const v8h*)(qrow + ka + 8));
    }

    float mrow[8], lrow[8];
    v8f o[4];
    #pragma unroll
    for (int r = 0; r < 8; ++r) { mrow[r] = -__builtin_inff(); lrow[r] = 0.f; }
    #pragma unroll
    for (int t = 0; t < 4; ++t)
        #pragma unroll
        for (int r = 0; r < 8; ++r) o[t][r] = 0.f;

    half_t* pl = lds + wave * (16 * 32);
    const half_t* krows = Kh + bh * SEQ * DK;
    const half_t* vbase = Vt + bh * DK * SEQ;

    for (int j = 0; j < SEQ; j += 32) {
        // ---- scores: two 16(q) x 16(k) tiles over d = 0..63 ----
        v8f s0, s1;
        #pragma unroll
        for (int r = 0; r < 8; ++r) { s0[r] = 0.f; s1[r] = 0.f; }
        #pragma unroll
        for (int f = 0; f < 2; ++f) {
            const int kb = f * 32 + (hi ? 16 : 0);
            v16h b0 = *(const v16h*)(krows + (size_t)(j + l15) * DK + kb);
            v16h b1 = *(const v16h*)(krows + (size_t)(j + 16 + l15) * DK + kb);
            s0 = __builtin_amdgcn_wmma_f32_16x16x32_f16(false, qa[f], false, b0,
                                                        (short)0, s0, false, false);
            s1 = __builtin_amdgcn_wmma_f32_16x16x32_f16(false, qa[f], false, b1,
                                                        (short)0, s1, false, false);
        }

        // ---- scale + threshold mask + online softmax (per q-row) ----
        float corr[8];
        #pragma unroll
        for (int r = 0; r < 8; ++r) {
            float a0 = s0[r] * 0.125f; a0 = (a0 <= thr) ? NEGINF : a0;
            float a1 = s1[r] * 0.125f; a1 = (a1 <= thr) ? NEGINF : a1;
            float tmax = fmaxf(a0, a1);
            #pragma unroll
            for (int off = 1; off < 16; off <<= 1)
                tmax = fmaxf(tmax, __shfl_xor(tmax, off, 32));
            const float mnew = fmaxf(mrow[r], tmax);
            const float p0 = __expf(a0 - mnew);
            const float p1 = __expf(a1 - mnew);
            float rs = p0 + p1;
            #pragma unroll
            for (int off = 1; off < 16; off <<= 1)
                rs += __shfl_xor(rs, off, 32);
            corr[r] = __expf(mrow[r] - mnew);
            lrow[r] = lrow[r] * corr[r] + rs;
            mrow[r] = mnew;
            // stage P (C-layout -> row-major f16 in LDS)
            const int prow = r + (hi ? 8 : 0);
            pl[prow * 32 + l15]      = (half_t)p0;
            pl[prow * 32 + 16 + l15] = (half_t)p1;
        }
        #pragma unroll
        for (int t = 0; t < 4; ++t)
            #pragma unroll
            for (int r = 0; r < 8; ++r) o[t][r] *= corr[r];

        // cross-lane LDS transpose: ensure stores land before reads
        asm volatile("s_wait_dscnt 0" ::: "memory");

        // ---- read P back as A fragment (16q x 32k) ----
        const int pk = hi ? 16 : 0;
        v16h pa = cat8(*(const v8h*)(pl + l15 * 32 + pk),
                       *(const v8h*)(pl + l15 * 32 + pk + 8));

        // ---- O += P @ V ; V^T layout gives contiguous B-operand loads ----
        #pragma unroll
        for (int t = 0; t < 4; ++t) {
            const int dn  = t * 16 + l15;
            const int kb2 = j + (hi ? 16 : 0);
            v16h vb = *(const v16h*)(vbase + (size_t)dn * SEQ + kb2);
            o[t] = __builtin_amdgcn_wmma_f32_16x16x32_f16(false, pa, false, vb,
                                                          (short)0, o[t], false, false);
        }
    }

    // ---- normalize and write merged-head f16 [B,S,H*DK] ----
    #pragma unroll
    for (int r = 0; r < 8; ++r) {
        const int q = q0 + r + (hi ? 8 : 0);
        const float inv = 1.0f / lrow[r];
        #pragma unroll
        for (int t = 0; t < 4; ++t) {
            const int d = t * 16 + l15;
            O[((size_t)(b * SEQ) + q) * D_MODEL + h * DK + d] = (half_t)(o[t][r] * inv);
        }
    }
}

// ---------------------------------------------------------------------------
extern "C" void kernel_launch(void* const* d_in, const int* in_sizes, int n_in,
                              void* d_out, int out_size, void* d_ws, size_t ws_size,
                              hipStream_t stream) {
    (void)in_sizes; (void)n_in; (void)out_size; (void)ws_size;
    const float* q   = (const float*)d_in[0];
    const float* k   = (const float*)d_in[1];
    const float* v   = (const float*)d_in[2];
    const float* Wq  = (const float*)d_in[3];
    const float* bq  = (const float*)d_in[4];
    const float* Wk  = (const float*)d_in[5];
    const float* bk  = (const float*)d_in[6];
    const float* Wv  = (const float*)d_in[7];
    const float* bv  = (const float*)d_in[8];
    const float* Wo  = (const float*)d_in[9];
    const float* bo  = (const float*)d_in[10];
    const float* thr = (const float*)d_in[11];

    const size_t NX = (size_t)MROWS * D_MODEL;    // 4M elems
    const size_t NW = (size_t)D_MODEL * D_MODEL;  // 1M elems
    half_t* ws  = (half_t*)d_ws;
    half_t* qx  = ws; ws += NX;
    half_t* kx  = ws; ws += NX;
    half_t* vx  = ws; ws += NX;
    half_t* wqh = ws; ws += NW;
    half_t* wkh = ws; ws += NW;
    half_t* wvh = ws; ws += NW;
    half_t* woh = ws; ws += NW;
    half_t* Qh  = ws; ws += NX;   // [B,H,S,DK]
    half_t* Kh  = ws; ws += NX;   // [B,H,S,DK]
    half_t* Vth = ws; ws += NX;   // [B,H,DK,S]
    half_t* Ah  = ws; ws += NX;   // [B,S,H*DK]

    const int CT = 256;
    cvt_f32_f16<<<(int)((NX + CT - 1) / CT), CT, 0, stream>>>(q,  qx,  (int)NX);
    cvt_f32_f16<<<(int)((NX + CT - 1) / CT), CT, 0, stream>>>(k,  kx,  (int)NX);
    cvt_f32_f16<<<(int)((NX + CT - 1) / CT), CT, 0, stream>>>(v,  vx,  (int)NX);
    cvt_f32_f16<<<(int)((NW + CT - 1) / CT), CT, 0, stream>>>(Wq, wqh, (int)NW);
    cvt_f32_f16<<<(int)((NW + CT - 1) / CT), CT, 0, stream>>>(Wk, wkh, (int)NW);
    cvt_f32_f16<<<(int)((NW + CT - 1) / CT), CT, 0, stream>>>(Wv, wvh, (int)NW);
    cvt_f32_f16<<<(int)((NW + CT - 1) / CT), CT, 0, stream>>>(Wo, woh, (int)NW);

    dim3 gg(MROWS / 16, D_MODEL / GN_BLOCK);
    gemm_xwT_kernel<<<gg, 256, 0, stream>>>(qx, wqh, bq, Qh,  nullptr, D_MODEL, D_MODEL, 0);
    gemm_xwT_kernel<<<gg, 256, 0, stream>>>(kx, wkh, bk, Kh,  nullptr, D_MODEL, D_MODEL, 0);
    gemm_xwT_kernel<<<gg, 256, 0, stream>>>(vx, wvh, bv, Vth, nullptr, D_MODEL, D_MODEL, 1);

    dim3 ga(SEQ / 64, NHEAD, BATCH);
    attn_kernel<<<ga, 128, 0, stream>>>(Qh, Kh, Vth, thr, Ah);

    gemm_xwT_kernel<<<gg, 256, 0, stream>>>(Ah, woh, bo, nullptr, (float*)d_out,
                                            D_MODEL, D_MODEL, 2);
}
